// SampleNet_9612136808753
// MI455X (gfx1250) — compile-verified, matrix operand order checked by reading, not run on hardware
//
#include <hip/hip_runtime.h>
#include <cmath>
#include <stdint.h>

// ---------------------------------------------------------------------------
// SampleNet forward on MI455X (gfx1250, wave32, WMMA + async global->LDS).
// Activations live as f16 in [b][n][channel] layout so that all inter-layer
// staging traffic moves via per-lane GLOBAL_LOAD_ASYNC_TO_LDS_B128 copies
// (ASYNCcnt / s_wait_asynccnt), with BN+ReLU applied by a cheap in-LDS pass.
// Weights are pre-packed to f16 (zero-padded K) so W tiles are async too.
// ---------------------------------------------------------------------------

typedef __attribute__((ext_vector_type(16))) _Float16 v16h;
typedef __attribute__((ext_vector_type(8)))  _Float16 v8h;
typedef __attribute__((ext_vector_type(8)))  float    v8f;

#define BATCH 256
#define NPTS  1024
#define KSEL  32

// Issue one per-lane 16B async copy global->LDS (GV addressing).
__device__ __forceinline__ void async_copy_b128(uint32_t lds_addr, const void* gptr) {
  asm volatile("global_load_async_to_lds_b128 %0, %1, off"
               :: "v"(lds_addr), "v"((uint64_t)(uintptr_t)gptr)
               : "memory");
}
__device__ __forceinline__ void wait_async0() {
  asm volatile("s_wait_asynccnt 0x0" ::: "memory");
}
// Flat pointers to LDS carry the LDS byte address in the low 32 bits.
__device__ __forceinline__ uint32_t lds_addr_of(const void* p) {
  return (uint32_t)(uintptr_t)p;
}

// Build a 16-bit WMMA operand fragment from an LDS row (32 consecutive K
// values at `p`). Per CDNA5 ISA layout for 16-bit A/B 16x32:
//   element t of group g (g=0,1):  K = 16*g + 8*half + t   (half = lane>>4)
__device__ __forceinline__ v16h make_frag(const _Float16* p, int half) {
  v8h lo = *(const v8h*)(p + 8 * half);
  v8h hi = *(const v8h*)(p + 16 + 8 * half);
  v16h r;
#pragma unroll
  for (int t = 0; t < 8; ++t) { r[t] = lo[t]; r[8 + t] = hi[t]; }
  return r;
}

// ---------------------------------------------------------------------------
// 1x1 conv as GEMM. X: f16 [B][N][Ci]; Y: f16 [B][N][Co] (raw, pre-BN).
// Xhat = relu(scale*X + shift) applied in-LDS when apply_bn.
// Per-channel sum/sumsq accumulated online for this layer's BN.
// Block: 128 threads (4 waves). Grid: (NPTS/64, BATCH).
// ---------------------------------------------------------------------------
__global__ __launch_bounds__(128)
void conv_gemm(const _Float16* __restrict__ X, const float* __restrict__ scl,
               const float* __restrict__ sft, int apply_bn,
               const _Float16* __restrict__ Wh,   // packed f16 [Co][Kpad]
               const float* __restrict__ bias,
               _Float16* __restrict__ Y, float* __restrict__ sum,
               float* __restrict__ sumsq, int Ci, int Co, int Kpad) {
  // +8 half pitch: fragment reads stride 68 dwords across lanes -> no 64-bank
  // conflicts; b128 LDS load alignment preserved.
  __shared__ _Float16 Wlds[128 * 136];
  __shared__ _Float16 Xlds[64 * 136];
  const int b = blockIdx.y;
  const int n0 = blockIdx.x * 64;
  const int tid = threadIdx.x;
  const int P = Kpad + 8;

  // ---- Stage W [Co][Kpad] via async DMA (pure byte copy, pre-packed f16).
  {
    const int lr = Kpad >> 3;  // 16B segments per row
    for (int idx = tid; idx < Co * lr; idx += 128) {
      int co = idx / lr, seg = idx % lr;
      async_copy_b128(lds_addr_of(&Wlds[co * P]) + seg * 16,
                      Wh + (size_t)co * Kpad + seg * 8);
    }
  }

  // ---- Stage X tile [n][k].
  if (Ci == Kpad) {
    // Rows are contiguous 128/256B in global -> async per-lane copies.
    const int lrx = Ci >> 3;
    for (int idx = tid; idx < 64 * lrx; idx += 128) {
      int n = idx / lrx, seg = idx % lrx;
      async_copy_b128(lds_addr_of(&Xlds[n * P]) + seg * 16,
                      X + ((size_t)b * NPTS + n0 + n) * Ci + seg * 8);
    }
    wait_async0();           // covers W and X DMAs of this wave
    __syncthreads();         // all waves' DMAs landed
    if (apply_bn) {          // in-LDS BN+ReLU of the previous layer
      for (int idx = tid; idx < 64 * Ci; idx += 128) {
        int n = idx / Ci, k = idx % Ci;
        float v = (float)Xlds[n * P + k];
        Xlds[n * P + k] = (_Float16)fmaxf(v * scl[k] + sft[k], 0.f);
      }
    }
  } else {
    // Layer 1: Ci=3 rows (6B) -> scalar staging with zero K-padding.
    for (int idx = tid; idx < 64 * Kpad; idx += 128) {
      int n = idx % 64, k = idx / 64;
      float v = (k < Ci) ? (float)X[((size_t)b * NPTS + n0 + n) * Ci + k] : 0.f;
      Xlds[n * P + k] = (_Float16)v;
    }
    wait_async0();           // W DMAs
  }
  __syncthreads();

  const int wave = tid >> 5, lane = tid & 31;
  const int half = lane >> 4, l15 = lane & 15;
  const int wn0 = wave * 16;           // this wave's 16-col strip
  const int ktiles = Kpad >> 5;

  for (int mt = 0; mt < (Co >> 4); ++mt) {
    v8f acc;
#pragma unroll
    for (int j = 0; j < 8; ++j) acc[j] = bias[mt * 16 + j + (half ? 8 : 0)];
    for (int kt = 0; kt < ktiles; ++kt) {
      v16h a = make_frag(&Wlds[(mt * 16 + l15) * P + kt * 32], half);
      v16h bb = make_frag(&Xlds[(wn0 + l15) * P + kt * 32], half);
      acc = __builtin_amdgcn_wmma_f32_16x16x32_f16(
          false, a, false, bb, (short)0, acc, false, false);
    }
    // Raw output: 8 contiguous channels per lane -> one packed 16B store.
    const int ng = n0 + wn0 + l15;
    v8h pk;
#pragma unroll
    for (int j = 0; j < 8; ++j) pk[j] = (_Float16)acc[j];
    *(v8h*)(Y + ((size_t)b * NPTS + ng) * Co + mt * 16 + 8 * half) = pk;
    // Online per-channel stats (f32).
#pragma unroll
    for (int j = 0; j < 8; ++j) {
      int Mg = mt * 16 + j + (half ? 8 : 0);
      float s1 = acc[j], s2 = acc[j] * acc[j];
#pragma unroll
      for (int m = 1; m < 16; m <<= 1) {
        s1 += __shfl_xor(s1, m, 32);
        s2 += __shfl_xor(s2, m, 32);
      }
      if (l15 == 0) {
        atomicAdd(&sum[Mg], s1);
        atomicAdd(&sumsq[Mg], s2);
      }
    }
  }
}

// ---------------------------------------------------------------------------
// FC layer: Y[r, o] = sum_k Xhat[r,k] * W[o,k] + bias[o], r in [0,256).
// W pre-packed f16 [O][I] -> async staging. X f32 (tiny) staged scalar.
// Block 128 threads (4 waves): 16 batch rows x 64 out channels per block.
// ---------------------------------------------------------------------------
__global__ __launch_bounds__(128)
void fc_gemm(const float* __restrict__ X, const float* __restrict__ scl,
             const float* __restrict__ sft, int apply_bn,
             const _Float16* __restrict__ Wh, const float* __restrict__ bias,
             float* __restrict__ Y, float* __restrict__ sum,
             float* __restrict__ sumsq, int collect_stats, int I, int O) {
  __shared__ _Float16 Xlds[16 * 264];
  __shared__ _Float16 Wlds[64 * 264];
  const int o0 = blockIdx.x * 64;
  const int r0 = blockIdx.y * 16;
  const int tid = threadIdx.x;
  const int P = I + 8;

  {
    const int lr = I >> 3;
    for (int idx = tid; idx < 64 * lr; idx += 128) {
      int o = idx / lr, seg = idx % lr;
      async_copy_b128(lds_addr_of(&Wlds[o * P]) + seg * 16,
                      Wh + (size_t)(o0 + o) * I + seg * 8);
    }
  }
  for (int idx = tid; idx < 16 * I; idx += 128) {
    int r = idx / I, k = idx % I;
    float v = X[(r0 + r) * I + k];
    if (apply_bn) v = fmaxf(v * scl[k] + sft[k], 0.f);
    Xlds[r * P + k] = (_Float16)v;
  }
  wait_async0();
  __syncthreads();

  const int wave = tid >> 5, lane = tid & 31;
  const int half = lane >> 4, l15 = lane & 15;
  const int og = o0 + wave * 16 + l15;   // this lane's output channel (N idx)
  const int ktiles = I >> 5;

  v8f acc;
#pragma unroll
  for (int j = 0; j < 8; ++j) acc[j] = bias[og];
  for (int kt = 0; kt < ktiles; ++kt) {
    v16h a = make_frag(&Xlds[l15 * P + kt * 32], half);                 // rows=batch
    v16h bb = make_frag(&Wlds[(wave * 16 + l15) * P + kt * 32], half);  // cols=out
    acc = __builtin_amdgcn_wmma_f32_16x16x32_f16(
        false, a, false, bb, (short)0, acc, false, false);
  }
  float s1 = 0.f, s2 = 0.f;
#pragma unroll
  for (int j = 0; j < 8; ++j) {
    int row = r0 + j + (half ? 8 : 0);
    Y[(size_t)row * O + og] = acc[j];
    s1 += acc[j];
    s2 += acc[j] * acc[j];
  }
  if (collect_stats) {
    s1 += __shfl_xor(s1, 16, 32);
    s2 += __shfl_xor(s2, 16, 32);
    if (half == 0) {
      atomicAdd(&sum[og], s1);
      atomicAdd(&sumsq[og], s2);
    }
  }
}

// Per-channel BN stats -> affine scale/shift (train-mode, biased variance).
__global__ void bn_finalize(const float* __restrict__ sum,
                            const float* __restrict__ sumsq,
                            const float* __restrict__ gamma,
                            const float* __restrict__ beta,
                            float* __restrict__ scale, float* __restrict__ shift,
                            int C, float invCount) {
  int c = blockIdx.x * blockDim.x + threadIdx.x;
  if (c >= C) return;
  float m = sum[c] * invCount;
  float v = sumsq[c] * invCount - m * m;
  float s = gamma[c] * rsqrtf(v + 1e-5f);
  scale[c] = s;
  shift[c] = beta[c] - m * s;
}

// relu(scale*y+shift) then max over N. Y layout [b][n][C], C==128.
// One block per batch row; thread t owns channel t&127, n-slice t>>7.
__global__ __launch_bounds__(256)
void bn_relu_maxpool(const _Float16* __restrict__ Y, const float* __restrict__ scl,
                     const float* __restrict__ sft, float* __restrict__ pooled,
                     int C) {
  const int b = blockIdx.x, tid = threadIdx.x;
  const int c = tid & (C - 1);
  const int slice = tid >> 7;
  const float s = scl[c], h = sft[c];
  float mx = 0.f;  // relu output >= 0
  for (int n = slice; n < NPTS; n += 2)
    mx = fmaxf(mx, fmaxf((float)Y[((size_t)b * NPTS + n) * C + c] * s + h, 0.f));
  __shared__ float red[256];
  red[tid] = mx;
  __syncthreads();
  if (tid < 128) pooled[b * C + tid] = fmaxf(red[tid], red[tid + 128]);
}

// W f32 [Co][Ci] -> f16 [Co][Kpad] zero-padded (Ci==Kpad for fc layers).
__global__ void pack_w(const float* __restrict__ W, _Float16* __restrict__ Wh,
                       int Ci, int Kpad, int total) {
  int i = blockIdx.x * 256 + threadIdx.x;
  if (i >= total) return;
  int co = i / Kpad, k = i % Kpad;
  Wh[i] = (_Float16)((k < Ci) ? W[co * Ci + k] : 0.f);
}

// x f32 [b][3][n] -> f16 [b][n][3].
__global__ void pack_x(const float* __restrict__ x, _Float16* __restrict__ xt) {
  int i = blockIdx.x * 256 + threadIdx.x;
  if (i >= BATCH * 3 * NPTS) return;
  int b = i / (3 * NPTS), r = i % (3 * NPTS);
  int c = r / NPTS, n = r % NPTS;
  xt[((size_t)b * NPTS + n) * 3 + c] = (_Float16)x[i];
}

// ---------------------------------------------------------------------------
// Fused sequential Gumbel-softmax top-k + einsum('bcn,bkn->bck').
// One 256-thread block per batch row; logits row and x row are LDS-resident.
// ---------------------------------------------------------------------------
__global__ __launch_bounds__(256)
void gumbel_topk(const float* __restrict__ logits, const float* __restrict__ x,
                 float* __restrict__ out) {
  __shared__ float lg[NPTS];
  __shared__ float xs[3 * NPTS];
  __shared__ float red[256];
  const int b = blockIdx.x, tid = threadIdx.x;

  for (int i = tid; i < NPTS; i += 256) lg[i] = logits[(size_t)b * NPTS + i];
  for (int i = tid; i < 3 * NPTS; i += 256) xs[i] = x[(size_t)b * 3 * NPTS + i];
  __syncthreads();

  float mx = -1e30f;
#pragma unroll
  for (int t = 0; t < 4; ++t) mx = fmaxf(mx, lg[tid + t * 256]);
  red[tid] = mx;
  __syncthreads();
  for (int s = 128; s > 0; s >>= 1) {
    if (tid < s) red[tid] = fmaxf(red[tid], red[tid + s]);
    __syncthreads();
  }
  float m0 = red[0];
  __syncthreads();
#pragma unroll
  for (int t = 0; t < 4; ++t) lg[tid + t * 256] -= m0;
  __syncthreads();

  for (int k = 0; k < KSEL; ++k) {
    float l4[4], mk = -1e30f;
#pragma unroll
    for (int t = 0; t < 4; ++t) { l4[t] = lg[tid + t * 256]; mk = fmaxf(mk, l4[t]); }
    red[tid] = mk;
    __syncthreads();
    for (int s = 128; s > 0; s >>= 1) {
      if (tid < s) red[tid] = fmaxf(red[tid], red[tid + s]);
      __syncthreads();
    }
    mk = red[0];
    __syncthreads();

    float e4[4], ps = 0.f;
#pragma unroll
    for (int t = 0; t < 4; ++t) { e4[t] = expf(l4[t] - mk); ps += e4[t]; }
    red[tid] = ps;
    __syncthreads();
    for (int s = 128; s > 0; s >>= 1) {
      if (tid < s) red[tid] += red[tid + s];
      __syncthreads();
    }
    float inv = 1.f / red[0];
    __syncthreads();

    float d0 = 0.f, d1 = 0.f, d2 = 0.f;
#pragma unroll
    for (int t = 0; t < 4; ++t) {
      int i = tid + t * 256;
      float z = e4[t] * inv;
      d0 += xs[i] * z;
      d1 += xs[NPTS + i] * z;
      d2 += xs[2 * NPTS + i] * z;
      lg[i] = l4[t] + logf(1.f - z + 1e-8f);
    }
    red[tid] = d0; __syncthreads();
    for (int s = 128; s > 0; s >>= 1) { if (tid < s) red[tid] += red[tid + s]; __syncthreads(); }
    d0 = red[0]; __syncthreads();
    red[tid] = d1; __syncthreads();
    for (int s = 128; s > 0; s >>= 1) { if (tid < s) red[tid] += red[tid + s]; __syncthreads(); }
    d1 = red[0]; __syncthreads();
    red[tid] = d2; __syncthreads();
    for (int s = 128; s > 0; s >>= 1) { if (tid < s) red[tid] += red[tid + s]; __syncthreads(); }
    d2 = red[0]; __syncthreads();

    if (tid == 0) {
      const int o = b * 96 + k;  // out[b][c][k], c stride 32
      out[o] = d0; out[o + 32] = d1; out[o + 64] = d2;
      out[24576 + o] = d0; out[24576 + o + 32] = d1; out[24576 + o + 64] = d2;
    }
    __syncthreads();
  }
}

// ---------------------------------------------------------------------------
extern "C" void kernel_launch(void* const* d_in, const int* in_sizes, int n_in,
                              void* d_out, int out_size, void* d_ws, size_t ws_size,
                              hipStream_t stream) {
  (void)in_sizes; (void)n_in; (void)out_size; (void)ws_size;
  // pytree leaf order: x, conv (w,b)x5, bn (g,be)x5, fc (w,b)x4, bn_fc (g,be)x3
  const float* x = (const float*)d_in[0];
  const float *cw[5], *cb[5], *cg[5], *cbe[5];
  for (int i = 0; i < 5; ++i) { cw[i] = (const float*)d_in[1 + 2 * i]; cb[i] = (const float*)d_in[2 + 2 * i]; }
  for (int i = 0; i < 5; ++i) { cg[i] = (const float*)d_in[11 + 2 * i]; cbe[i] = (const float*)d_in[12 + 2 * i]; }
  const float *fw[4], *fb[4], *fg[3], *fbe[3];
  for (int i = 0; i < 4; ++i) { fw[i] = (const float*)d_in[21 + 2 * i]; fb[i] = (const float*)d_in[22 + 2 * i]; }
  for (int i = 0; i < 3; ++i) { fg[i] = (const float*)d_in[29 + 2 * i]; fbe[i] = (const float*)d_in[30 + 2 * i]; }

  char* wsb = (char*)d_ws;
  size_t off = 0;
  auto carve = [&](size_t bytes) -> void* {
    void* p = wsb + off;
    off += (bytes + 255) & ~(size_t)255;
    return p;
  };
  _Float16* actA = (_Float16*)carve((size_t)BATCH * NPTS * 128 * 2);  // 64 MB
  _Float16* actB = (_Float16*)carve((size_t)BATCH * NPTS * 128 * 2);  // 64 MB
  _Float16* xt   = (_Float16*)carve((size_t)BATCH * NPTS * 3 * 2);
  float* csum   = (float*)carve(128 * 4);
  float* csq    = (float*)carve(128 * 4);
  float* cscale = (float*)carve(5 * 128 * 4);
  float* cshift = (float*)carve(5 * 128 * 4);
  float* pooled = (float*)carve((size_t)BATCH * 128 * 4);
  float* fcA    = (float*)carve((size_t)BATCH * 256 * 4);
  float* fcB    = (float*)carve((size_t)BATCH * 256 * 4);
  float* fsum   = (float*)carve(256 * 4);
  float* fsq    = (float*)carve(256 * 4);
  float* fscale = (float*)carve(3 * 256 * 4);
  float* fshift = (float*)carve(3 * 256 * 4);
  float* logits = (float*)carve((size_t)BATCH * NPTS * 4);

  // Packed f16 weights.
  const int cCi[5] = {3, 64, 64, 64, 128}, cCo[5] = {64, 64, 64, 128, 128},
            cKp[5] = {32, 64, 64, 64, 128};
  _Float16* cwh[5];
  for (int i = 0; i < 5; ++i) cwh[i] = (_Float16*)carve((size_t)cCo[i] * cKp[i] * 2);
  const int fI[4] = {128, 256, 256, 256}, fO[4] = {256, 256, 256, 1024};
  _Float16* fwh[4];
  for (int i = 0; i < 4; ++i) fwh[i] = (_Float16*)carve((size_t)fO[i] * fI[4 == 4 ? i : i] * 2);

  const float invBN = 1.f / (float)(BATCH * NPTS);
  const float invB  = 1.f / (float)BATCH;
  const dim3 cgrid(NPTS / 64, BATCH);

  pack_x<<<(BATCH * 3 * NPTS + 255) / 256, 256, 0, stream>>>(x, xt);
  for (int i = 0; i < 5; ++i) {
    int tot = cCo[i] * cKp[i];
    pack_w<<<(tot + 255) / 256, 256, 0, stream>>>(cw[i], cwh[i], cCi[i], cKp[i], tot);
  }
  for (int i = 0; i < 4; ++i) {
    int tot = fO[i] * fI[i];
    pack_w<<<(tot + 255) / 256, 256, 0, stream>>>(fw[i], fwh[i], fI[i], fI[i], tot);
  }

  struct { const _Float16* in; _Float16* out; int Ci, Co, Kpad; } L[5] = {
      {xt,   actA, 3,   64,  32},
      {actA, actB, 64,  64,  64},
      {actB, actA, 64,  64,  64},
      {actA, actB, 64,  128, 64},
      {actB, actA, 128, 128, 128},
  };
  for (int i = 0; i < 5; ++i) {
    hipMemsetAsync(csum, 0, 128 * sizeof(float), stream);
    hipMemsetAsync(csq, 0, 128 * sizeof(float), stream);
    const float* s = (i == 0) ? cscale : cscale + (i - 1) * 128;
    const float* h = (i == 0) ? cshift : cshift + (i - 1) * 128;
    conv_gemm<<<cgrid, 128, 0, stream>>>(L[i].in, s, h, (i != 0), cwh[i], cb[i],
                                         L[i].out, csum, csq, L[i].Ci, L[i].Co,
                                         L[i].Kpad);
    bn_finalize<<<1, 128, 0, stream>>>(csum, csq, cg[i], cbe[i],
                                       cscale + i * 128, cshift + i * 128,
                                       L[i].Co, invBN);
  }

  bn_relu_maxpool<<<BATCH, 256, 0, stream>>>(actA, cscale + 4 * 128,
                                             cshift + 4 * 128, pooled, 128);

  struct { const float* in; float* out; int bn, stats; } F[4] = {
      {pooled, fcA,    0, 1},
      {fcA,    fcB,    1, 1},
      {fcB,    fcA,    1, 1},
      {fcA,    logits, 1, 0},
  };
  for (int i = 0; i < 4; ++i) {
    if (F[i].stats) {
      hipMemsetAsync(fsum, 0, 256 * sizeof(float), stream);
      hipMemsetAsync(fsq, 0, 256 * sizeof(float), stream);
    }
    const float* s = (i == 0) ? fscale : fscale + (i - 1) * 256;
    const float* h = (i == 0) ? fshift : fshift + (i - 1) * 256;
    fc_gemm<<<dim3(fO[i] / 64, BATCH / 16), 128, 0, stream>>>(
        F[i].in, s, h, F[i].bn, fwh[i], fb[i], F[i].out, fsum, fsq, F[i].stats,
        fI[i], fO[i]);
    if (F[i].stats)
      bn_finalize<<<2, 128, 0, stream>>>(fsum, fsq, fg[i], fbe[i],
                                         fscale + i * 256, fshift + i * 256,
                                         fO[i], invB);
  }

  gumbel_topk<<<BATCH, 256, 0, stream>>>(logits, x, (float*)d_out);
}
